// COVtorch_68101001445527
// MI455X (gfx1250) — compile-verified, matrix operand order checked by reading, not run on hardware
//
#include <hip/hip_runtime.h>

// MI455X (gfx1250) — wave32, WMMA.
// Per-batch covariance via hi/lo bf16-split Gram matrix with v_wmma_f32_16x16x32_bf16.
// Roofline: 409.6 MB input @ 23.3 TB/s => ~17.6us floor. bf16 WMMA (3x split ~ 39 GFLOP)
// stays near that floor; f32 WMMA (K=4) would be compute-bound ~5x slower.

typedef __attribute__((ext_vector_type(16))) __bf16 v16bf;
typedef __attribute__((ext_vector_type(8)))  float  v8f;

#define N_OBS 50000
#define D_FEAT 64
#define OUT_F 256
#define N_BATCH 32
#define WAVES_PER_BATCH 64
#define CHUNK 800   // ceil(50000/64) rounded up to multiple of 32

// ---------------------------------------------------------------- zero ws
__global__ void zero_ws_kernel(float* __restrict__ ws, int n) {
    int i = blockIdx.x * blockDim.x + threadIdx.x;
    if (i < n) ws[i] = 0.0f;
}

// Build one 16-wide feature block's A/B operand (identical per-lane contents for
// both the A (16x32) and B (32x16) roles of the symmetric Gram GEMM) in hi/lo
// bf16 split form.  p points at x[b][n][I*16 + lm].  Element t of the v16bf
// holds observation kk = hf*8 + t (t<8) or hf*8 + 8 + t (t>=8), matching the
// 16-bit 16x32 A layout (lanes 0-15: K=0..7,16..23; lanes 16-31: K=8..15,24..31).
// REM is a compile-time constant in the hot path so the compare folds away.
template <bool GUARDED>
__device__ __forceinline__ void build_ops(const float* __restrict__ p, int hf, int rem,
                                          v16bf& hi, v16bf& lo, float& sum) {
#pragma unroll
    for (int t = 0; t < 16; ++t) {
        int kk = hf * 8 + ((t < 8) ? t : (8 + t));
        float v = (!GUARDED || kk < rem) ? p[kk * D_FEAT] : 0.0f;
        sum += v;
        __bf16 h = (__bf16)v;
        hi[t] = h;
        lo[t] = (__bf16)(v - (float)h);
    }
}

// One 32-observation step: 4 operand blocks (hi+lo) feed all 10 upper-triangle
// tiles; 30 v_wmma_f32_16x16x32_bf16 per step.
__device__ __forceinline__ void wmma_step(const v16bf hi[4], const v16bf lo[4], v8f acc[10]) {
    int t = 0;
#pragma unroll
    for (int I = 0; I < 4; ++I) {
#pragma unroll
        for (int J = I; J < 4; ++J, ++t) {
            acc[t] = __builtin_amdgcn_wmma_f32_16x16x32_bf16(
                false, hi[I], false, hi[J], (short)0, acc[t], false, false);
            acc[t] = __builtin_amdgcn_wmma_f32_16x16x32_bf16(
                false, hi[I], false, lo[J], (short)0, acc[t], false, false);
            acc[t] = __builtin_amdgcn_wmma_f32_16x16x32_bf16(
                false, lo[I], false, hi[J], (short)0, acc[t], false, false);
        }
    }
}

// ---------------------------------------------------------------- accumulate
// grid.x = N_BATCH * 8 blocks, 256 threads = 8 waves; 64 independent waves/batch.
__global__ void __launch_bounds__(256)
cov_accum_kernel(const float* __restrict__ x,
                 float* __restrict__ S,      // [32][64][64], upper-tri blocks valid
                 float* __restrict__ sums)   // [32][64]
{
    const int lane  = threadIdx.x & 31;
    const int wave  = threadIdx.x >> 5;
    const int batch = blockIdx.x >> 3;
    const int wib   = ((blockIdx.x & 7) << 3) + wave;   // 0..63
    const int lm = lane & 15;
    const int hf = lane >> 4;

    const int n0     = wib * CHUNK;
    const int nLimit = min(n0 + CHUNK, N_OBS);
    // n0 and CHUNK are multiples of 32 -> at most one partial (tail) step.
    const int nFull  = n0 + ((nLimit > n0) ? ((nLimit - n0) & ~31) : 0);

    const float* xb = x + (size_t)batch * N_OBS * D_FEAT;

    v8f acc[10] = {};            // upper-triangle tiles (I<=J), f32 accumulators
    float sloc[4] = {0.f, 0.f, 0.f, 0.f};

    // ---- branch-free main loop (rem compare folds: kk<=31 < 32) ----
    for (int n = n0; n < nFull; n += 32) {
        const float* p0 = xb + (size_t)n * D_FEAT + lm;
        v16bf hi[4], lo[4];
#pragma unroll
        for (int I = 0; I < 4; ++I)
            build_ops<false>(p0 + I * 16, hf, 32, hi[I], lo[I], sloc[I]);
        // prefetch next 32-observation slab -> global_prefetch_b8
        __builtin_prefetch(p0 + 32 * D_FEAT, 0, 1);
        wmma_step(hi, lo, acc);
    }

    // ---- single guarded tail step (only last partial wave takes it) ----
    if (nFull < nLimit) {
        const int rem = nLimit - nFull;              // 1..31
        const float* p0 = xb + (size_t)nFull * D_FEAT + lm;
        v16bf hi[4], lo[4];
#pragma unroll
        for (int I = 0; I < 4; ++I)
            build_ops<true>(p0 + I * 16, hf, rem, hi[I], lo[I], sloc[I]);
        wmma_step(hi, lo, acc);
    }

    // ---- fold partials into global accumulators ----
#pragma unroll
    for (int I = 0; I < 4; ++I)
        atomicAdd(&sums[batch * D_FEAT + I * 16 + lm], sloc[I]);

    // C/D layout: VGPR r -> row M = r + hf*8, col = lm
    int t = 0;
#pragma unroll
    for (int I = 0; I < 4; ++I) {
#pragma unroll
        for (int J = I; J < 4; ++J, ++t) {
#pragma unroll
            for (int r = 0; r < 8; ++r) {
                int M = r + hf * 8;
                atomicAdd(&S[((size_t)batch * D_FEAT + I * 16 + M) * D_FEAT + J * 16 + lm],
                          acc[t][r]);
            }
        }
    }
}

// ---------------------------------------------------------------- finalize
// grid.x = 32 (one block per batch), 256 threads (one per output feature).
__global__ void __launch_bounds__(256)
finalize_kernel(const float* __restrict__ S,
                const float* __restrict__ sums,
                const float* __restrict__ W,
                const float* __restrict__ bias,
                float* __restrict__ out)
{
    __shared__ __align__(16) float cov[D_FEAT * D_FEAT];
    __shared__ float red[OUT_F];

    const int b   = blockIdx.x;
    const int tid = threadIdx.x;
    const float invN   = 1.0f / (float)N_OBS;
    const float invNm1 = 1.0f / (float)(N_OBS - 1);

    const float* Sb = S + (size_t)b * D_FEAT * D_FEAT;
    const float* sb = sums + b * D_FEAT;

    // Build symmetric covariance in LDS; cov = (S - s s^T / N) / (N-1).
    // Only upper-tri 16x16 blocks of S were written; mirror at read time.
    for (int e = tid; e < D_FEAT * D_FEAT; e += OUT_F) {
        int i = e >> 6, j = e & 63;
        float s = ((i >> 4) <= (j >> 4)) ? Sb[e] : Sb[j * D_FEAT + i];
        cov[e] = (s - sb[i] * sb[j] * invN) * invNm1;
    }
    __syncthreads();

    // out[b][tid] = bias[tid] + cov_flat . W[tid]
    const float4* w4 = (const float4*)(W + (size_t)tid * (D_FEAT * D_FEAT));
    const float4* c4 = (const float4*)cov;
    float a = bias[tid];
    for (int q = 0; q < (D_FEAT * D_FEAT) / 4; ++q) {
        float4 wv = w4[q];
        float4 cv = c4[q];
        a = fmaf(cv.x, wv.x, a);
        a = fmaf(cv.y, wv.y, a);
        a = fmaf(cv.z, wv.z, a);
        a = fmaf(cv.w, wv.w, a);
    }

    // L2 normalize across the 256 outputs of this batch.
    red[tid] = a * a;
    __syncthreads();
#pragma unroll
    for (int s = 128; s > 0; s >>= 1) {
        if (tid < s) red[tid] += red[tid + s];
        __syncthreads();
    }
    float norm = fmaxf(sqrtf(red[0]), 1e-12f);
    out[b * OUT_F + tid] = a / norm;
}

// ---------------------------------------------------------------- launch
extern "C" void kernel_launch(void* const* d_in, const int* in_sizes, int n_in,
                              void* d_out, int out_size, void* d_ws, size_t ws_size,
                              hipStream_t stream) {
    const float* x    = (const float*)d_in[0];   // [32][50000][64] f32
    const float* W    = (const float*)d_in[1];   // [256][4096] f32
    const float* bias = (const float*)d_in[2];   // [256] f32
    float* out = (float*)d_out;                  // [32][256] f32

    float* S    = (float*)d_ws;                              // 32*4096 floats
    float* sums = S + (size_t)N_BATCH * D_FEAT * D_FEAT;     // 32*64 floats
    const int ztot = N_BATCH * D_FEAT * D_FEAT + N_BATCH * D_FEAT;

    zero_ws_kernel<<<(ztot + 255) / 256, 256, 0, stream>>>(S, ztot);
    cov_accum_kernel<<<N_BATCH * 8, 256, 0, stream>>>(x, S, sums);
    finalize_kernel<<<N_BATCH, 256, 0, stream>>>(S, sums, W, bias, out);
}
